// ChannelAttention_9079560863765
// MI455X (gfx1250) — matrix operand order repaired and checked by reading, not compiled, not run on hardware
//
#include <hip/hip_runtime.h>
#include <hip/hip_bf16.h>

typedef __attribute__((ext_vector_type(16))) __bf16 v16bf;
typedef __attribute__((ext_vector_type(8)))  __bf16 v8bf;
typedef __attribute__((ext_vector_type(8)))  float  v8f;
typedef __attribute__((ext_vector_type(4)))  unsigned int v4u;
typedef __attribute__((ext_vector_type(8)))  int v8i_t;
typedef __attribute__((ext_vector_type(4)))  int v4i_t;

union FragAB {
    v16bf v;
    v8bf  h[2];
    __bf16 e[16];
};

#define CC    512
#define NN    4096
#define C8C   64
#define BB    16

__device__ __forceinline__ v8f wmma_bf16(v16bf a, v16bf b, v8f c) {
    return __builtin_amdgcn_wmma_f32_16x16x32_bf16(false, a, false, b,
                                                   (short)0, c, false, false);
}

// ---------------------------------------------------------------------------
// Kernel 0: one-time weight conversion f32 -> bf16 workspace.
// wqkv = [Wq;Wk;Wv] stacked [192][512], wo = [512][64]. 64*512 == 512*64.
// ---------------------------------------------------------------------------
__global__ __launch_bounds__(256) void cvt_weights_kernel(
    const float* __restrict__ Wq, const float* __restrict__ Wk,
    const float* __restrict__ Wv, const float* __restrict__ Wo,
    __bf16* __restrict__ wqkv, __bf16* __restrict__ wo)
{
    const int i = blockIdx.x * 256 + threadIdx.x;   // grid covers 32768
    const int WQK = C8C * CC;                        // 32768
    wqkv[i]           = (__bf16)Wq[i];
    wqkv[WQK + i]     = (__bf16)Wk[i];
    wqkv[2 * WQK + i] = (__bf16)Wv[i];
    wo[i]             = (__bf16)Wo[i];
}

// ---------------------------------------------------------------------------
// Kernel 1: q/k/v = W{q,k,v} @ x[b] + bias -> bf16 [3][B][64][N]
// grid (B, N/128), 256 threads (8 waves). Wave w owns n-tile w, all 12 m-tiles.
// Every x element is loaded+converted exactly once per block.
// ---------------------------------------------------------------------------
__global__ __launch_bounds__(256) void qkv_kernel(
    const float* __restrict__ x,
    const __bf16* __restrict__ wqkv,     // [192][512] bf16
    const float* __restrict__ bq, const float* __restrict__ bk,
    const float* __restrict__ bv,
    __bf16* __restrict__ qkv)
{
    const int b    = blockIdx.x;
    const int n0   = blockIdx.y * 128;
    const int lane = threadIdx.x & 31;
    const int wave = threadIdx.x >> 5;      // n-tile index 0..7
    const int lh   = lane >> 4;
    const int lm   = lane & 15;
    const int col  = n0 + wave * 16 + lm;

    const float* __restrict__ xb = x + (size_t)b * CC * NN + col;

    v8f acc[12];
    #pragma unroll
    for (int t = 0; t < 12; ++t) acc[t] = (v8f)0.f;

    for (int ks = 0; ks < 16; ++ks) {        // K steps of 32 over C=512
        const int k0 = ks * 32;

        FragAB bf;                           // B fragment (x), 8 cvt_pk/lane
        const int krow = k0 + lh * 16;
        #pragma unroll
        for (int i = 0; i < 16; ++i)
            bf.e[i] = (__bf16)xb[(krow + i) * NN];

        #pragma unroll
        for (int t = 0; t < 12; ++t) {       // A fragments: raw bf16 b128 loads
            const __bf16* __restrict__ wr =
                wqkv + (t * 16 + lm) * CC + k0 + lh * 16;
            FragAB a;
            a.h[0] = *(const v8bf*)(wr);
            a.h[1] = *(const v8bf*)(wr + 8);
            acc[t] = wmma_bf16(a.v, bf.v, acc[t]);
        }
    }

    #pragma unroll
    for (int t = 0; t < 12; ++t) {
        const int mat = t >> 2;
        const int mtl = t & 3;
        const float* __restrict__ bias =
            (mat == 0) ? bq : ((mat == 1) ? bk : bv);
        #pragma unroll
        for (int r = 0; r < 8; ++r) {
            const int ch = mtl * 16 + lh * 8 + r;
            const float val = acc[t][r] + bias[ch];
            qkv[(((size_t)mat * BB + b) * C8C + ch) * NN + col] = (__bf16)val;
        }
    }
}

// ---------------------------------------------------------------------------
// Kernel 2: energy = q @ k^T (64x64, K=4096), softmax -> attn bf16 [B][64][64]
// ---------------------------------------------------------------------------
__global__ __launch_bounds__(256) void attn_kernel(
    const __bf16* __restrict__ qkv,
    __bf16* __restrict__ attn)
{
    __shared__ float e_s[64][64];

    const int b    = blockIdx.x;
    const int lane = threadIdx.x & 31;
    const int wave = threadIdx.x >> 5;
    const int lh   = lane >> 4;
    const int lm   = lane & 15;

    const __bf16* __restrict__ q = qkv + (size_t)b * C8C * NN;
    const __bf16* __restrict__ k = qkv + (size_t)(BB + b) * C8C * NN;

    const int mt  = wave >> 1;
    const int ntb = (wave & 1) * 2;

    v8f acc[2];
    acc[0] = (v8f)0.f; acc[1] = (v8f)0.f;

    const __bf16* __restrict__ qrow = q + (mt * 16 + lm) * NN;

    for (int ks = 0; ks < 128; ++ks) {
        const int k0 = ks * 32;
        FragAB a;
        a.h[0] = *(const v8bf*)(qrow + k0 + lh * 16);
        a.h[1] = *(const v8bf*)(qrow + k0 + 8 + lh * 16);
        #pragma unroll
        for (int j = 0; j < 2; ++j) {
            const __bf16* __restrict__ krow = k + ((ntb + j) * 16 + lm) * NN;
            const v16bf bvv = *(const v16bf*)(krow + k0 + lh * 16);
            acc[j] = wmma_bf16(a.v, bvv, acc[j]);
        }
    }

    #pragma unroll
    for (int j = 0; j < 2; ++j)
        #pragma unroll
        for (int r = 0; r < 8; ++r)
            e_s[mt * 16 + lh * 8 + r][(ntb + j) * 16 + lm] = acc[j][r];
    __syncthreads();

    if (threadIdx.x < 64) {
        const int row = threadIdx.x;
        float mx = -1e30f;
        for (int c = 0; c < 64; ++c) mx = fmaxf(mx, e_s[row][c]);
        float s = 0.f;
        for (int c = 0; c < 64; ++c) s += __expf(e_s[row][c] - mx);
        const float inv = 1.f / s;
        for (int c = 0; c < 64; ++c)
            attn[((size_t)b * 64 + row) * 64 + c] =
                (__bf16)(__expf(e_s[row][c] - mx) * inv);
    }
}

// ---------------------------------------------------------------------------
// Kernel 3: tmp = attn @ v_tile (v tile staged into LDS by the Tensor Data
// Mover), then out_tile = Wo @ tmp + bo.  grid (B, N/128), 256 threads.
// ---------------------------------------------------------------------------
__global__ __launch_bounds__(256) void out_kernel(
    const __bf16* __restrict__ qkv,
    const __bf16* __restrict__ attn,
    const __bf16* __restrict__ wo,       // [512][64] bf16
    const float* __restrict__ bo,
    float* __restrict__ out)
{
    __shared__ __bf16 vsl[C8C * 128];    // v tile [64 ch][128 col] (16KB)
    __shared__ __bf16 ts[C8C * 128];     // tmp staging [64][128] (16KB)

    const int b    = blockIdx.x;
    const int n0   = blockIdx.y * 128;
    const int lane = threadIdx.x & 31;
    const int wave = threadIdx.x >> 5;
    const int lh   = lane >> 4;
    const int lm   = lane & 15;

    const __bf16* __restrict__ v  = qkv + (size_t)(2 * BB + b) * C8C * NN;
    const __bf16* __restrict__ am = attn + (size_t)b * C8C * C8C;

    // ---- stage v tile into LDS ----
#if __has_builtin(__builtin_amdgcn_tensor_load_to_lds)
    if (wave == 0) {
        // Tensor DMA descriptor (D#), 2D tensor: dim0 = spatial (contig),
        // dim1 = channel (stride 4096). Tile = 128 x 64, bf16 (data_size=1).
        const unsigned lds_addr = (unsigned)(size_t)(&vsl[0]);
        const unsigned long long ga = (unsigned long long)(size_t)(v + n0);
        v4u g0;
        g0[0] = 1u;                                   // count=1 (valid user D#)
        g0[1] = lds_addr;                             // lds_addr[31:0]
        g0[2] = (unsigned)(ga & 0xFFFFFFFFu);         // global_addr[31:0]
        g0[3] = (unsigned)((ga >> 32) & 0x01FFFFFFu)  // global_addr[56:32]
              | 0x80000000u;                          // type=2 ("image")
        v8i_t g1;
        g1[0] = 0x00010000;          // data_size=1 (2 bytes/elem)
        g1[1] = (int)((NN & 0xFFFF) << 16);   // tensor_dim0[15:0] in [31:16]
        g1[2] = (int)((NN >> 16) | (C8C << 16)); // dim0 hi | tensor_dim1 lo
        g1[3] = (int)(128 << 16);    // tensor_dim1 hi=0 | tile_dim0=128
        g1[4] = C8C;                 // tile_dim1=64, tile_dim2=0
        g1[5] = NN;                  // tensor_dim0_stride[31:0] = 4096
        g1[6] = 0;                   // stride0 hi | stride1 lo
        g1[7] = 0;
        v4i_t gz4 = {0, 0, 0, 0};
        v8i_t gz8 = {0, 0, 0, 0, 0, 0, 0, 0};
        __builtin_amdgcn_tensor_load_to_lds(g0, g1, gz4, gz4, gz8, 0);
        __builtin_amdgcn_s_wait_tensorcnt(0);
    }
#else
    for (int i = threadIdx.x; i < 1024; i += 256) {   // 8192 elems / 8
        const int ch = i >> 4, off = (i & 15) * 8;
        *(v8bf*)(vsl + ch * 128 + off) =
            *(const v8bf*)(v + ch * NN + n0 + off);
    }
#endif
    __syncthreads();

    // ---- stage A: tmp = attn(64x64) @ v_tile(64x128) ----
    const int mt  = wave >> 1;
    const int ntb = (wave & 1) * 4;
    v8f acc[4];
    #pragma unroll
    for (int j = 0; j < 4; ++j) acc[j] = (v8f)0.f;

    const __bf16* __restrict__ arow = am + (mt * 16 + lm) * C8C;
    #pragma unroll
    for (int ks = 0; ks < 2; ++ks) {
        const int k0 = ks * 32;
        FragAB a;
        a.h[0] = *(const v8bf*)(arow + k0 + lh * 16);
        a.h[1] = *(const v8bf*)(arow + k0 + 8 + lh * 16);
        #pragma unroll
        for (int j = 0; j < 4; ++j) {
            const int colL = (ntb + j) * 16 + lm;
            FragAB bb;
            #pragma unroll
            for (int i = 0; i < 16; ++i)
                bb.e[i] = vsl[(k0 + lh * 16 + i) * 128 + colL];
            acc[j] = wmma_bf16(a.v, bb.v, acc[j]);
        }
    }
    #pragma unroll
    for (int j = 0; j < 4; ++j)
        #pragma unroll
        for (int r = 0; r < 8; ++r)
            ts[(mt * 16 + lh * 8 + r) * 128 + (ntb + j) * 16 + lm] =
                (__bf16)acc[j][r];
    __syncthreads();

    // ---- stage B: out_tile = Wo(512x64) @ tmp(64x128) + bo ----
    #pragma unroll
    for (int m4 = 0; m4 < 4; ++m4) {
        const int mt2 = wave * 4 + m4;                  // 0..31
        const __bf16* __restrict__ wrow = wo + (mt2 * 16 + lm) * C8C;
        FragAB a0, a1;
        a0.h[0] = *(const v8bf*)(wrow + 0  + lh * 16);
        a0.h[1] = *(const v8bf*)(wrow + 8  + lh * 16);
        a1.h[0] = *(const v8bf*)(wrow + 32 + lh * 16);
        a1.h[1] = *(const v8bf*)(wrow + 40 + lh * 16);
        for (int nt = 0; nt < 8; ++nt) {
            FragAB b0, b1;
            #pragma unroll
            for (int i = 0; i < 16; ++i)
                b0.e[i] = ts[(lh * 16 + i) * 128 + nt * 16 + lm];
            #pragma unroll
            for (int i = 0; i < 16; ++i)
                b1.e[i] = ts[(32 + lh * 16 + i) * 128 + nt * 16 + lm];
            v8f c = (v8f)0.f;
            c = wmma_bf16(a0.v, b0.v, c);
            c = wmma_bf16(a1.v, b1.v, c);
            #pragma unroll
            for (int r = 0; r < 8; ++r) {
                const int row = mt2 * 16 + lh * 8 + r;
                out[((size_t)b * CC + row) * NN + n0 + nt * 16 + lm] =
                    c[r] + bo[row];
            }
        }
    }
}

// ---------------------------------------------------------------------------
extern "C" void kernel_launch(void* const* d_in, const int* in_sizes, int n_in,
                              void* d_out, int out_size, void* d_ws, size_t ws_size,
                              hipStream_t stream) {
    const float* x  = (const float*)d_in[0];
    const float* Wq = (const float*)d_in[1];
    const float* bq = (const float*)d_in[2];
    const float* Wk = (const float*)d_in[3];
    const float* bk = (const float*)d_in[4];
    const float* Wv = (const float*)d_in[5];
    const float* bv = (const float*)d_in[6];
    const float* Wo = (const float*)d_in[7];
    const float* bo = (const float*)d_in[8];
    float* out = (float*)d_out;

    __bf16* qkv  = (__bf16*)d_ws;                          // 3*B*64*N
    __bf16* attn = qkv + (size_t)3 * BB * C8C * NN;        // B*64*64
    __bf16* wqkv = attn + (size_t)BB * C8C * C8C;          // 192*512
    __bf16* wo   = wqkv + (size_t)3 * C8C * CC;            // 512*64

    dim3 blk(256);
    cvt_weights_kernel<<<dim3(128), blk, 0, stream>>>(Wq, Wk, Wv, Wo, wqkv, wo);
    qkv_kernel<<<dim3(BB, NN / 128), blk, 0, stream>>>(x, wqkv, bq, bk, bv, qkv);
    attn_kernel<<<dim3(BB), blk, 0, stream>>>(qkv, attn);
    out_kernel<<<dim3(BB, NN / 128), blk, 0, stream>>>(qkv, attn, wo, bo, out);
}
